// CollaborativePerceptionGNN_74594991997476
// MI455X (gfx1250) — compile-verified, matrix operand order checked by reading, not run on hardware
//
#include <hip/hip_runtime.h>
#include <hip/hip_bf16.h>
#include <math.h>

#define N_NODES 20000
#define N_EDGES 60000
#define F_INDIM 16
#define H 64
#define LAYERS 3
#define G_GRAPHS 64
#define EPS 1e-5f

typedef _Float16 f16;
typedef _Float16 v16h __attribute__((ext_vector_type(16)));
typedef _Float16 h8   __attribute__((ext_vector_type(8)));
typedef float    v8f  __attribute__((ext_vector_type(8)));

#if defined(__gfx1250__) && __has_builtin(__builtin_amdgcn_global_load_async_to_lds_b128) && __has_builtin(__builtin_amdgcn_s_wait_asynccnt)
#define USE_ASYNC_LDS 1
#else
#define USE_ASYNC_LDS 0
#endif

static __device__ __forceinline__ v8f wmma16(v16h a, v16h b, v8f c) {
    // D = A(16x32 f16) * B(32x16 f16) + C(16x16 f32)
    return __builtin_amdgcn_wmma_f32_16x16x32_f16(false, a, false, b, (short)0, c, false, false);
}

#if USE_ASYNC_LDS
// builtin signature (from hipcc diagnostic): first arg is
// 'int __attribute__((vector_size(16))) __device__ *' (global/AS1 int4*),
// second is the LDS-side (AS3) int4*; last two are immediates (offset, cpol).
typedef int i4v __attribute__((vector_size(4 * sizeof(int))));
typedef __attribute__((address_space(1))) i4v* as1_i4p;
typedef __attribute__((address_space(3))) i4v* as3_i4p;

// async DMA of one 16B packet global -> LDS (ASYNCcnt-tracked, no VGPR staging)
static __device__ __forceinline__ void async_cp16(const h8* g, h8* l_) {
    __builtin_amdgcn_global_load_async_to_lds_b128(
        (as1_i4p)(void*)(g),
        (as3_i4p)(void*)(l_), 0, 0);
}
#endif

__device__ __forceinline__ unsigned fkey(float f) {
    unsigned b = __float_as_uint(f);
    return (b & 0x80000000u) ? ~b : (b | 0x80000000u);
}
__device__ __forceinline__ float funkey(unsigned k) {
    unsigned b = (k & 0x80000000u) ? (k & 0x7fffffffu) : ~k;
    return __uint_as_float(b);
}

// ---------------- utility kernels ----------------
__global__ void k_zero(float* __restrict__ p, int n) {
    int i = blockIdx.x * blockDim.x + threadIdx.x;
    if (i < n) p[i] = 0.0f;
}

__global__ void k_deg(const int* __restrict__ edge_index, float* __restrict__ deg) {
    int e = blockIdx.x * blockDim.x + threadIdx.x;
    if (e < N_EDGES) atomicAdd(&deg[edge_index[N_EDGES + e]], 1.0f);
}

__global__ void k_invdeg(const float* __restrict__ deg, float* __restrict__ invd) {
    int i = blockIdx.x * blockDim.x + threadIdx.x;
    if (i < N_NODES) {
        float d = deg[i];
        invd[i] = d > 0.0f ? 1.0f / d : 0.0f;
    }
}

// h = x @ emb_W + emb_b ; also write f16 copy
__global__ void k_emb(const float* __restrict__ x, const float* __restrict__ emb_W,
                      const float* __restrict__ emb_b,
                      float* __restrict__ h, f16* __restrict__ hh) {
    int idx = blockIdx.x * blockDim.x + threadIdx.x;   // exact N*H
    int node = idx >> 6, ch = idx & (H - 1);
    float acc = emb_b[ch];
#pragma unroll
    for (int i = 0; i < F_INDIM; ++i)
        acc += x[node * F_INDIM + i] * emb_W[i * H + ch];
    h[idx] = acc;
    hh[idx] = (f16)acc;
}

// Bt[((l*64+i)*64+o)*32+k] = edge_W2[l][k][i*64+o]   (f16)
__global__ void k_cvt_W2(const float* __restrict__ W2, f16* __restrict__ Bt) {
    int idx = blockIdx.x * blockDim.x + threadIdx.x;   // exact 3*64*64*32
    int k = idx & 31, o = (idx >> 5) & 63, i = (idx >> 11) & 63, l = idx >> 17;
    Bt[idx] = (f16)W2[((size_t)(l * 32 + k)) * 4096 + i * 64 + o];
}

// B2t[((l*2+s)*64+o)*32+k] = edge_b2[l][(s*32+k)*64+o]
__global__ void k_cvt_b2(const float* __restrict__ b2, f16* __restrict__ B2t) {
    int idx = blockIdx.x * blockDim.x + threadIdx.x;   // exact 3*2*64*32
    int k = idx & 31, o = (idx >> 5) & 63, s = (idx >> 11) & 1, l = idx >> 12;
    B2t[idx] = (f16)b2[(size_t)l * 4096 + (s * 32 + k) * 64 + o];
}

// SWt[((l*2+s)*64+o)*32+k] = self_W[l][s*32+k][o]
__global__ void k_cvt_selfW(const float* __restrict__ sw, f16* __restrict__ SWt) {
    int idx = blockIdx.x * blockDim.x + threadIdx.x;   // exact 3*2*64*32
    int k = idx & 31, o = (idx >> 5) & 63, s = (idx >> 11) & 1, l = idx >> 12;
    SWt[idx] = (f16)sw[((size_t)l * 64 + s * 32 + k) * 64 + o];
}

// ---------------- fused ECC edge kernel (the hot loop) ----------------
// One wave per 16-edge tile; block = 10 waves. K = 64 i-steps (x32 hid) + 2 bias steps.
// B' chunks double-buffered in LDS; async global->LDS DMA overlaps WMMA when available.
__global__ __launch_bounds__(320)
void k_edge(int l, const float* __restrict__ h, const f16* __restrict__ hh,
            const int* __restrict__ edge_index,
            const float* __restrict__ edge_W1, const float* __restrict__ edge_b1,
            const f16* __restrict__ Bt, const f16* __restrict__ B2t,
            float* __restrict__ agg) {
    __shared__ h8 sB[2][2048];  // 2 x 32 KB: 8 i-steps x 64 o x 32 k (f16)
    const int tid = threadIdx.x;
    const int wave = tid >> 5, lane = tid & 31;
    const int m = lane & 15, half = lane >> 4, kbase = half * 8;
    const int tile = blockIdx.x * 10 + wave;
    const int e = tile * 16 + m;

    const int r = edge_index[e];
    const int c = edge_index[N_EDGES + e];

    // edge_attr = ||h[row,:3]-h[col,:3]||
    float d0 = h[(size_t)r * H + 0] - h[(size_t)c * H + 0];
    float d1 = h[(size_t)r * H + 1] - h[(size_t)c * H + 1];
    float d2 = h[(size_t)r * H + 2] - h[(size_t)c * H + 2];
    float ea = sqrtf(d0 * d0 + d1 * d1 + d2 * d2);

    // this lane's 16-slot hid fragment: hid[k] = relu(ea*W1[k]+b1[k])
    const float* W1 = edge_W1 + l * 32;
    const float* b1 = edge_b1 + l * 32;
    v16h hidv;
#pragma unroll
    for (int j = 0; j < 16; ++j) {
        int k = (j < 8) ? (kbase + j) : (16 + kbase + (j - 8));
        hidv[j] = (f16)fmaxf(ea * W1[k] + b1[k], 0.0f);
    }

    // h[row] row in f16 (64 halves)
    h8 hrv[8];
    const h8* hp = (const h8*)(hh + (size_t)r * H);
#pragma unroll
    for (int t = 0; t < 8; ++t) hrv[t] = hp[t];

    v8f acc[4] = {};
    const h8* Btv = (const h8*)Bt + (size_t)l * 64 * 256;  // layer base (h8 units; 256 per i-step)

    // ---- prologue: stage chunk 0 into buffer 0 ----
#if USE_ASYNC_LDS
#pragma unroll
    for (int j = 0; j < 7; ++j) {
        int idx = tid + j * 320;
        if (idx < 2048) async_cp16(Btv + idx, &sB[0][idx]);
    }
    __builtin_amdgcn_s_wait_asynccnt(0);
    __syncthreads();
#else
    for (int idx = tid; idx < 2048; idx += 320) sB[0][idx] = Btv[idx];
    __syncthreads();
#endif

#pragma unroll
    for (int ci = 0; ci < 8; ++ci) {
        const int p = ci & 1;
#if USE_ASYNC_LDS
        // kick off DMA for the next chunk; it overlaps the 32 WMMAs below
        if (ci < 7) {
#pragma unroll
            for (int j = 0; j < 7; ++j) {
                int idx = tid + j * 320;
                if (idx < 2048) async_cp16(Btv + (ci + 1) * 2048 + idx, &sB[1 - p][idx]);
            }
        }
#else
        h8 stage[7];
        if (ci < 7) {
#pragma unroll
            for (int j = 0; j < 7; ++j) {
                int idx = tid + j * 320;
                if (idx < 2048) stage[j] = Btv[(ci + 1) * 2048 + idx];
            }
        }
#endif
        const f16* sBh = (const f16*)&sB[p][0];
#pragma unroll
        for (int ii = 0; ii < 8; ++ii) {
            const int i = ci * 8 + ii;
            f16 s = hrv[i >> 3][i & 7];     // h[row][i]
            v16h a;
#pragma unroll
            for (int j = 0; j < 16; ++j) a[j] = (f16)(s * hidv[j]);  // outer-product A frag
#pragma unroll
            for (int nt = 0; nt < 4; ++nt) {
                const int o = nt * 16 + m;
                const int off = (ii * 64 + o) * 32 + kbase;
                h8 blo = *(const h8*)(sBh + off);
                h8 bhi = *(const h8*)(sBh + off + 16);
                v16h b;
#pragma unroll
                for (int j = 0; j < 8; ++j) { b[j] = blo[j]; b[j + 8] = bhi[j]; }
                acc[nt] = wmma16(a, b, acc[nt]);
            }
        }
#if USE_ASYNC_LDS
        if (ci < 7) __builtin_amdgcn_s_wait_asynccnt(0);
        __syncthreads();
#else
        if (ci < 7) {
#pragma unroll
            for (int j = 0; j < 7; ++j) {
                int idx = tid + j * 320;
                if (idx < 2048) sB[1 - p][idx] = stage[j];
            }
        }
        __syncthreads();
#endif
    }

    // edge_b2 contribution: msg += h[row] @ B2'  (2 extra K-steps, B from L2)
#pragma unroll
    for (int s2 = 0; s2 < 2; ++s2) {
        v16h a;
#pragma unroll
        for (int j = 0; j < 16; ++j) {
            int k = (j < 8) ? (kbase + j) : (16 + kbase + (j - 8));
            int i = s2 * 32 + k;
            a[j] = hrv[i >> 3][i & 7];
        }
#pragma unroll
        for (int nt = 0; nt < 4; ++nt) {
            const int o = nt * 16 + m;
            const size_t off = (((size_t)l * 2 + s2) * 64 + o) * 32 + kbase;
            h8 blo = *(const h8*)(B2t + off);
            h8 bhi = *(const h8*)(B2t + off + 16);
            v16h b;
#pragma unroll
            for (int j = 0; j < 8; ++j) { b[j] = blo[j]; b[j + 8] = bhi[j]; }
            acc[nt] = wmma16(a, b, acc[nt]);
        }
    }

    // scatter: lane holds rows M=v+8*half (edges), channel o=nt*16+m
#pragma unroll
    for (int v = 0; v < 8; ++v) {
        const int em = tile * 16 + v + 8 * half;
        const int dst = edge_index[N_EDGES + em];
#pragma unroll
        for (int nt = 0; nt < 4; ++nt)
            atomicAdd(&agg[(size_t)dst * H + nt * 16 + m], acc[nt][v]);
    }
}

// ---------------- self-GEMM + mean-agg + BN-stats ----------------
__global__ __launch_bounds__(320)
void k_node(int l, const f16* __restrict__ hh, const f16* __restrict__ SWt,
            const float* __restrict__ agg, const float* __restrict__ invd,
            const float* __restrict__ self_b,
            float* __restrict__ out_pre, float* __restrict__ stats) {
    __shared__ h8 sB[512];  // 8 KB: [s:2][o:64][k:32]
    const int tid = threadIdx.x;
    const int wave = tid >> 5, lane = tid & 31;
    const int m = lane & 15, half = lane >> 4, kbase = half * 8;
    const int tile = blockIdx.x * 10 + wave;

    const h8* SWv = (const h8*)SWt;
    for (int idx = tid; idx < 512; idx += 320) sB[idx] = SWv[(size_t)l * 512 + idx];
    __syncthreads();

    const int n0 = tile * 16 + m;
    h8 hrv[8];
    const h8* hp = (const h8*)(hh + (size_t)n0 * H);
#pragma unroll
    for (int t = 0; t < 8; ++t) hrv[t] = hp[t];

    v8f acc[4] = {};
    const f16* sBh = (const f16*)sB;
#pragma unroll
    for (int s = 0; s < 2; ++s) {
        v16h a;
#pragma unroll
        for (int j = 0; j < 16; ++j) {
            int k = (j < 8) ? (kbase + j) : (16 + kbase + (j - 8));
            int i = s * 32 + k;
            a[j] = hrv[i >> 3][i & 7];
        }
#pragma unroll
        for (int nt = 0; nt < 4; ++nt) {
            const int o = nt * 16 + m;
            const int off = (s * 64 + o) * 32 + kbase;
            h8 blo = *(const h8*)(sBh + off);
            h8 bhi = *(const h8*)(sBh + off + 16);
            v16h b;
#pragma unroll
            for (int j = 0; j < 8; ++j) { b[j] = blo[j]; b[j + 8] = bhi[j]; }
            acc[nt] = wmma16(a, b, acc[nt]);
        }
    }

#pragma unroll
    for (int nt = 0; nt < 4; ++nt) {
        const int ch = nt * 16 + m;
        const float bs = self_b[l * H + ch];
        float ssum = 0.f, ssq = 0.f;
#pragma unroll
        for (int v = 0; v < 8; ++v) {
            const int node = tile * 16 + v + 8 * half;
            float val = agg[(size_t)node * H + ch] * invd[node] + acc[nt][v] + bs;
            out_pre[(size_t)node * H + ch] = val;
            ssum += val; ssq += val * val;
        }
        atomicAdd(&stats[ch], ssum);
        atomicAdd(&stats[H + ch], ssq);
    }
}

__global__ void k_bn_final(const float* __restrict__ stats, float* __restrict__ mu_rs) {
    int c = threadIdx.x;
    float mu = stats[c] * (1.0f / (float)N_NODES);
    float var = stats[H + c] * (1.0f / (float)N_NODES) - mu * mu;
    mu_rs[c] = mu;
    mu_rs[H + c] = rsqrtf(var + EPS);
}

__global__ void k_apply(int l, const float* __restrict__ out_pre,
                        const float* __restrict__ mu_rs,
                        const float* __restrict__ gma, const float* __restrict__ bta,
                        float* __restrict__ h, f16* __restrict__ hh) {
    int idx = blockIdx.x * blockDim.x + threadIdx.x;   // exact N*H
    int ch = idx & (H - 1);
    float val = out_pre[idx];
    val = gma[l * H + ch] * (val - mu_rs[ch]) * mu_rs[H + ch] + bta[l * H + ch];
    val = fmaxf(val, 0.0f);
    float nh = h[idx] + val;
    h[idx] = nh;
    hh[idx] = (f16)nh;
}

// ---------------- pooling + classifier ----------------
__global__ void k_pool_init(float* __restrict__ gsum, unsigned* __restrict__ gmaxu,
                            float* __restrict__ gcnt) {
    int idx = blockIdx.x * blockDim.x + threadIdx.x;
    if (idx < G_GRAPHS * H) { gsum[idx] = 0.0f; gmaxu[idx] = 0u; }
    if (idx < G_GRAPHS) gcnt[idx] = 0.0f;
}

__global__ void k_pool(const float* __restrict__ h, const int* __restrict__ batch,
                       float* __restrict__ gsum, unsigned* __restrict__ gmaxu,
                       float* __restrict__ gcnt) {
    int idx = blockIdx.x * blockDim.x + threadIdx.x;   // exact N*H
    int node = idx >> 6, ch = idx & (H - 1);
    int g = batch[node];
    float v = h[idx];
    atomicAdd(&gsum[g * H + ch], v);
    atomicMax(&gmaxu[g * H + ch], fkey(v));
    if (ch == 0) atomicAdd(&gcnt[g], 1.0f);
}

__global__ void k_cls(const float* __restrict__ gsum, const unsigned* __restrict__ gmaxu,
                      const float* __restrict__ gcnt,
                      const float* __restrict__ W1, const float* __restrict__ b1,
                      const float* __restrict__ W2, const float* __restrict__ b2,
                      float* __restrict__ out) {
    int g = threadIdx.x;   // 64 threads, one graph each
    float cnt = gcnt[g];
    float gvec[2 * H];
    for (int ch = 0; ch < H; ++ch) {
        gvec[ch] = (cnt > 0.f) ? gsum[g * H + ch] / cnt : 0.0f;
        gvec[H + ch] = (cnt > 0.f) ? funkey(gmaxu[g * H + ch]) : 0.0f;
    }
    float logit = b2[0];
    for (int j = 0; j < H; ++j) {
        float a = b1[j];
        for (int i = 0; i < 2 * H; ++i) a += gvec[i] * W1[i * H + j];
        logit += fmaxf(a, 0.0f) * W2[j];
    }
    out[g] = 1.0f / (1.0f + expf(-logit));
}

// ---------------- launcher ----------------
extern "C" void kernel_launch(void* const* d_in, const int* in_sizes, int n_in,
                              void* d_out, int out_size, void* d_ws, size_t ws_size,
                              hipStream_t stream) {
    const float* x        = (const float*)d_in[0];
    const int*   eidx     = (const int*)d_in[1];
    const int*   batch    = (const int*)d_in[2];
    const float* emb_W    = (const float*)d_in[3];
    const float* emb_b    = (const float*)d_in[4];
    const float* edge_W1  = (const float*)d_in[5];
    const float* edge_b1  = (const float*)d_in[6];
    const float* edge_W2  = (const float*)d_in[7];
    const float* edge_b2  = (const float*)d_in[8];
    const float* self_W   = (const float*)d_in[9];
    const float* self_b   = (const float*)d_in[10];
    const float* bn_gamma = (const float*)d_in[11];
    const float* bn_beta  = (const float*)d_in[12];
    const float* cls_W1   = (const float*)d_in[13];
    const float* cls_b1   = (const float*)d_in[14];
    const float* cls_W2   = (const float*)d_in[15];
    const float* cls_b2   = (const float*)d_in[16];
    float* out = (float*)d_out;

    char* ws = (char*)d_ws;
    size_t off = 0;
    auto alloc = [&](size_t bytes) -> void* {
        void* p = ws + off;
        off = (off + bytes + 255) & ~(size_t)255;
        return p;
    };
    const int NH = N_NODES * H;
    float*    h_buf  = (float*)alloc((size_t)NH * 4);
    f16*      hh_buf = (f16*)alloc((size_t)NH * 2);
    float*    agg    = (float*)alloc((size_t)NH * 4);   // stats directly follows (NH*4 % 256 == 0)
    float*    stats  = (float*)alloc(128 * 4);
    float*    out_pre = (float*)alloc((size_t)NH * 4);
    float*    mu_rs  = (float*)alloc(128 * 4);
    float*    deg    = (float*)alloc((size_t)N_NODES * 4);
    float*    invd   = (float*)alloc((size_t)N_NODES * 4);
    f16*      Bt     = (f16*)alloc((size_t)LAYERS * 64 * 64 * 32 * 2);
    f16*      B2t    = (f16*)alloc((size_t)LAYERS * 2 * 64 * 32 * 2);
    f16*      SWt    = (f16*)alloc((size_t)LAYERS * 2 * 64 * 32 * 2);
    float*    gsum   = (float*)alloc((size_t)G_GRAPHS * H * 4);
    unsigned* gmaxu  = (unsigned*)alloc((size_t)G_GRAPHS * H * 4);
    float*    gcnt   = (float*)alloc((size_t)G_GRAPHS * 4);

    // degrees
    k_zero<<<(N_NODES + 255) / 256, 256, 0, stream>>>(deg, N_NODES);
    k_deg<<<(N_EDGES + 255) / 256, 256, 0, stream>>>(eidx, deg);
    k_invdeg<<<(N_NODES + 255) / 256, 256, 0, stream>>>(deg, invd);

    // embedding + weight conversions (f16, WMMA fragment layouts)
    k_emb<<<NH / 256, 256, 0, stream>>>(x, emb_W, emb_b, h_buf, hh_buf);
    k_cvt_W2<<<(LAYERS * 64 * 64 * 32) / 256, 256, 0, stream>>>(edge_W2, Bt);
    k_cvt_b2<<<(LAYERS * 2 * 64 * 32) / 256, 256, 0, stream>>>(edge_b2, B2t);
    k_cvt_selfW<<<(LAYERS * 2 * 64 * 32) / 256, 256, 0, stream>>>(self_W, SWt);

    for (int l = 0; l < LAYERS; ++l) {
        k_zero<<<(NH + 128 + 255) / 256, 256, 0, stream>>>(agg, NH + 128);  // agg + stats
        k_edge<<<N_EDGES / 16 / 10, 320, 0, stream>>>(l, h_buf, hh_buf, eidx,
                                                      edge_W1, edge_b1, Bt, B2t, agg);
        k_node<<<N_NODES / 16 / 10, 320, 0, stream>>>(l, hh_buf, SWt, agg, invd,
                                                      self_b, out_pre, stats);
        k_bn_final<<<1, 64, 0, stream>>>(stats, mu_rs);
        k_apply<<<NH / 256, 256, 0, stream>>>(l, out_pre, mu_rs, bn_gamma, bn_beta,
                                              h_buf, hh_buf);
    }

    k_pool_init<<<(G_GRAPHS * H + 255) / 256, 256, 0, stream>>>(gsum, gmaxu, gcnt);
    k_pool<<<NH / 256, 256, 0, stream>>>(h_buf, batch, gsum, gmaxu, gcnt);
    k_cls<<<1, 64, 0, stream>>>(gsum, gmaxu, gcnt, cls_W1, cls_b1, cls_W2, cls_b2, out);
}
